// CausalSelfAttention_29944511988372
// MI455X (gfx1250) — matrix operand order repaired and checked by reading, not compiled, not run on hardware
//
#include <hip/hip_runtime.h>

// ---------------- problem constants ----------------
#define BB 2
#define TT 2048
#define DD 1024
#define HH 16
#define DHD 64

typedef __attribute__((ext_vector_type(16))) __bf16 v16bf;
typedef __attribute__((ext_vector_type(8)))  float  v8f;
typedef __attribute__((ext_vector_type(8)))  unsigned int v8u;

__device__ __forceinline__ unsigned short f2bf(float f) {
  unsigned int u = __float_as_uint(f);
  unsigned int r = u + 0x7FFFu + ((u >> 16) & 1u);   // round-to-nearest-even
  return (unsigned short)(r >> 16);
}
__device__ __forceinline__ unsigned int pack2bf(float lo, float hi) {
  return (unsigned int)f2bf(lo) | ((unsigned int)f2bf(hi) << 16);
}
__device__ __forceinline__ int imin(int a, int b) { return a < b ? a : b; }

// -------- gfx1250 async global->LDS copy (ASYNCcnt path), guarded --------
#if __has_builtin(__builtin_amdgcn_global_load_async_to_lds_b128)
#define ASYNC_LDS 1
typedef __attribute__((__vector_size__(4 * sizeof(int)))) int vi4;
typedef __attribute__((address_space(1))) vi4* gvi4p;
typedef __attribute__((address_space(3))) vi4* lvi4p;
__device__ __forceinline__ void async_b128(const void* g, void* l) {
  __builtin_amdgcn_global_load_async_to_lds_b128((gvi4p)g, (lvi4p)l, 0, 0);
}
__device__ __forceinline__ void wait_async0() {
  asm volatile("s_wait_asynccnt 0" ::: "memory");
}
#else
#define ASYNC_LDS 0
#endif

// ---------------- fp32 -> bf16 conversion ----------------
__global__ void cvt_f32_bf16_kernel(const float* __restrict__ src,
                                    unsigned short* __restrict__ dst, int n4) {
  int i = blockIdx.x * blockDim.x + threadIdx.x;
  if (i < n4) {
    float4 f = ((const float4*)src)[i];
    uint2 p;
    p.x = pack2bf(f.x, f.y);
    p.y = pack2bf(f.z, f.w);
    ((uint2*)dst)[i] = p;
  }
}

// ---------------- bf16 WMMA GEMM: C[M x 1024] = A[M x 1024] * B[1024 x 1024] ----------------
// SPLIT=true : write bf16 in (B,H,T,DH) head-split layout (for Q/K/V)
// SPLIT=false: write fp32 row-major (final projection to d_out)
template <bool SPLIT>
__global__ __launch_bounds__(256) void gemm_bf16_kernel(
    const unsigned short* __restrict__ A, const unsigned short* __restrict__ Bw,
    void* __restrict__ Out) {
  constexpr int N = DD, K = DD, BM = 128, BN = 128, BK = 32;
  __shared__ __attribute__((aligned(16))) unsigned short As[2][BM * BK];
  __shared__ __attribute__((aligned(16))) unsigned short Bs[2][BK * BN];

  const int tid  = threadIdx.x;
  const int lane = tid & 31, wave = tid >> 5;
  const int wm = wave >> 1, wn = wave & 1;          // 4 waves along M, 2 along N
  const int blockM = blockIdx.y * BM, blockN = blockIdx.x * BN;
  const int half = lane >> 4, l16 = lane & 15;

  v8f acc[2][4];
#pragma unroll
  for (int mt = 0; mt < 2; ++mt)
#pragma unroll
    for (int nt = 0; nt < 4; ++nt)
      acc[mt][nt] = (v8f){0.f, 0.f, 0.f, 0.f, 0.f, 0.f, 0.f, 0.f};

  const int arow = tid >> 1, acol = (tid & 1) * 16;  // A tile: 128x32, 16 bf16/thread
  const int brow = tid >> 3, bcol = (tid & 7) * 16;  // B tile: 32x128, 16 bf16/thread
  const unsigned short* Agt = A + (size_t)(blockM + arow) * K + acol;   // + k0
  const unsigned short* Bgt = Bw + (size_t)brow * N + blockN + bcol;    // + k0*N
  unsigned short* Asl0 = &As[0][arow * BK + acol];
  unsigned short* Bsl0 = &Bs[0][brow * BN + bcol];

#if ASYNC_LDS
  // -------- double-buffered pipeline fed by global_load_async_to_lds --------
  {
    async_b128(Agt, Asl0);
    async_b128(Agt + 8, Asl0 + 8);
    async_b128(Bgt, Bsl0);
    async_b128(Bgt + 8, Bsl0 + 8);
  }
  int buf = 0;
  for (int k0 = 0; k0 < K; k0 += BK) {
    wait_async0();        // my slice of buffer `buf` has landed in LDS
    __syncthreads();      // everyone's slice landed; everyone done reading buf^1
    if (k0 + BK < K) {
      const unsigned short* Ag = Agt + (k0 + BK);
      const unsigned short* Bg = Bgt + (size_t)(k0 + BK) * N;
      unsigned short* Al = Asl0 + (buf ^ 1) * (BM * BK);
      unsigned short* Bl = Bsl0 + (buf ^ 1) * (BK * BN);
      async_b128(Ag, Al);
      async_b128(Ag + 8, Al + 8);
      async_b128(Bg, Bl);
      async_b128(Bg + 8, Bl + 8);
    }
    const unsigned short* Asb = As[buf];
    const unsigned short* Bsb = Bs[buf];
#else
  // -------- fallback: synchronous staging --------
  for (int k0 = 0; k0 < K; k0 += BK) {
    const uint4* ag = (const uint4*)(Agt + k0);
    uint4 av0 = ag[0], av1 = ag[1];
    const uint4* bg = (const uint4*)(Bgt + (size_t)k0 * N);
    uint4 bv0 = bg[0], bv1 = bg[1];
    ((uint4*)Asl0)[0] = av0;
    ((uint4*)Asl0)[1] = av1;
    ((uint4*)Bsl0)[0] = bv0;
    ((uint4*)Bsl0)[1] = bv1;
    __syncthreads();
    const unsigned short* Asb = As[0];
    const unsigned short* Bsb = Bs[0];
#endif

    // A fragments (A 16x32 in contiguous k order: lanes<16 K={0..15}, lanes>=16 K={16..31})
    v16bf afrag[2];
#pragma unroll
    for (int mt = 0; mt < 2; ++mt) {
      v8u t;
      const unsigned short* ab = Asb + (wm * 32 + mt * 16 + l16) * BK;
#pragma unroll
      for (int j = 0; j < 8; ++j) {
        int kk = 2 * j + half * 16;
        t[j] = *(const unsigned int*)(ab + kk);
      }
      afrag[mt] = __builtin_bit_cast(v16bf, t);
    }
    // B fragments (lanes<16: cols, K=2j,2j+1; lanes>=16: K=16+2j,17+2j)
    v16bf bfrag[4];
#pragma unroll
    for (int nt = 0; nt < 4; ++nt) {
      v8u t;
      const unsigned short* bb = Bsb + wn * 64 + nt * 16 + l16;
#pragma unroll
      for (int j = 0; j < 8; ++j) {
        int kk = 2 * j + half * 16;
        unsigned int lo = bb[(size_t)kk * BN];
        unsigned int hi = bb[(size_t)(kk + 1) * BN];
        t[j] = (lo & 0xFFFFu) | (hi << 16);
      }
      bfrag[nt] = __builtin_bit_cast(v16bf, t);
    }

#pragma unroll
    for (int mt = 0; mt < 2; ++mt)
#pragma unroll
      for (int nt = 0; nt < 4; ++nt)
        acc[mt][nt] = __builtin_amdgcn_wmma_f32_16x16x32_bf16(
            false, afrag[mt], false, bfrag[nt], (short)0, acc[mt][nt], false, false);

#if ASYNC_LDS
    buf ^= 1;
  }
#else
    __syncthreads();
  }
#endif

  // store: C/D layout -> row M = r + 8*half, col N = l16
#pragma unroll
  for (int mt = 0; mt < 2; ++mt) {
#pragma unroll
    for (int nt = 0; nt < 4; ++nt) {
      int gRow0 = blockM + wm * 32 + mt * 16 + half * 8;
      int gCol  = blockN + wn * 64 + nt * 16 + l16;
#pragma unroll
      for (int r = 0; r < 8; ++r) {
        int row = gRow0 + r;
        float v = acc[mt][nt][r];
        if (SPLIT) {
          int b_ = row / TT, t_ = row % TT;
          int h_ = gCol >> 6, dh_ = gCol & 63;
          ((unsigned short*)Out)[(((size_t)b_ * HH + h_) * TT + t_) * DHD + dh_] = f2bf(v);
        } else {
          ((float*)Out)[(size_t)row * N + gCol] = v;
        }
      }
    }
  }
}

// ---------------- flash attention: one wave per (b,h, 16-row q-block) ----------------
__global__ __launch_bounds__(256) void attn_kernel(
    const unsigned short* __restrict__ Qb, const unsigned short* __restrict__ Kb,
    const unsigned short* __restrict__ Vb, unsigned short* __restrict__ Ob) {
  __shared__ float Pld[8][16 * 32];  // 2KB per wave scratch for P transpose
  const int tid = threadIdx.x, lane = tid & 31, wave = tid >> 5;
  const int task = blockIdx.x * 8 + wave;        // 4096 tasks
  const int qb = task & 127;                     // q-block index (16 rows each)
  const int bh = task >> 7;                      // b*H + h
  const size_t base = (size_t)bh * TT * DHD;
  const int half = lane >> 4, l16 = lane & 15;
  float* P = Pld[wave];

  // Q fragments: A-matrix 16x32 per dh-chunk (2 chunks cover DH=64)
  v16bf qf[2];
  {
    const int qrow = qb * 16 + l16;
    const unsigned int* qp = (const unsigned int*)(Qb + base + (size_t)qrow * DHD);
#pragma unroll
    for (int c = 0; c < 2; ++c) {
      v8u t;
#pragma unroll
      for (int j = 0; j < 8; ++j) {
        int dh = 2 * j + half * 16 + c * 32;
        t[j] = qp[dh >> 1];
      }
      qf[c] = __builtin_bit_cast(v16bf, t);
    }
  }

  v8f o[4];
#pragma unroll
  for (int nt = 0; nt < 4; ++nt) o[nt] = (v8f){0.f, 0.f, 0.f, 0.f, 0.f, 0.f, 0.f, 0.f};
  float mrow[8], lrow[8];
#pragma unroll
  for (int r = 0; r < 8; ++r) { mrow[r] = -1e30f; lrow[r] = 0.f; }

  const int kcount = qb * 16 + 16;
  for (int kb = 0; kb < kcount; kb += 32) {
    // S = Q * K^T for two 16-col tiles
    v8f s[2];
#pragma unroll
    for (int c = 0; c < 2; ++c) {
      const int kg = imin(kb + c * 16 + l16, TT - 1);
      const unsigned int* kp = (const unsigned int*)(Kb + base + (size_t)kg * DHD);
      v8f z = (v8f){0.f, 0.f, 0.f, 0.f, 0.f, 0.f, 0.f, 0.f};
#pragma unroll
      for (int ch = 0; ch < 2; ++ch) {
        v8u t;
#pragma unroll
        for (int j = 0; j < 8; ++j) {
          int dh = 2 * j + half * 16 + ch * 32;
          t[j] = kp[dh >> 1];
        }
        v16bf kf = __builtin_bit_cast(v16bf, t);
        z = __builtin_amdgcn_wmma_f32_16x16x32_bf16(false, qf[ch], false, kf,
                                                    (short)0, z, false, false);
      }
      s[c] = z;
    }

    // scale, causal mask, online softmax (row r+8*half spans 16 lanes of my half)
    float alpha[8];
#pragma unroll
    for (int r = 0; r < 8; ++r) {
      int row = qb * 16 + r + 8 * half;
      int c0 = kb + l16, c1 = kb + 16 + l16;
      float v0 = (c0 <= row) ? s[0][r] * 0.125f : -1e30f;
      float v1 = (c1 <= row) ? s[1][r] * 0.125f : -1e30f;
      float mx = fmaxf(v0, v1);
      mx = fmaxf(mx, __shfl_xor(mx, 1, 16));
      mx = fmaxf(mx, __shfl_xor(mx, 2, 16));
      mx = fmaxf(mx, __shfl_xor(mx, 4, 16));
      mx = fmaxf(mx, __shfl_xor(mx, 8, 16));
      float mnew = fmaxf(mrow[r], mx);
      alpha[r] = __expf(mrow[r] - mnew);
      mrow[r] = mnew;
      float p0 = __expf(v0 - mnew);
      float p1 = __expf(v1 - mnew);
      s[0][r] = p0; s[1][r] = p1;
      float ps = p0 + p1;
      ps += __shfl_xor(ps, 1, 16);
      ps += __shfl_xor(ps, 2, 16);
      ps += __shfl_xor(ps, 4, 16);
      ps += __shfl_xor(ps, 8, 16);
      lrow[r] = lrow[r] * alpha[r] + ps;
    }
#pragma unroll
    for (int nt = 0; nt < 4; ++nt)
#pragma unroll
      for (int r = 0; r < 8; ++r) o[nt][r] *= alpha[r];

    // transpose P (C layout -> A layout) through per-wave LDS
#pragma unroll
    for (int r = 0; r < 8; ++r) {
      int row = r + 8 * half;
      P[row * 32 + l16]      = s[0][r];
      P[row * 32 + 16 + l16] = s[1][r];
    }
    asm volatile("s_wait_dscnt 0" ::: "memory");  // CDNA5 split DS counter wait

    v16bf pf;
    {
      v8u t;
      const float* pr = P + l16 * 32;
#pragma unroll
      for (int j = 0; j < 8; ++j) {
        int kk = 2 * j + half * 16;
        t[j] = pack2bf(pr[kk], pr[kk + 1]);
      }
      pf = __builtin_bit_cast(v16bf, t);
    }

    // O += P * V  (B fragments from V, contraction over 32 keys)
#pragma unroll
    for (int nt = 0; nt < 4; ++nt) {
      v8u t;
      int dh = nt * 16 + l16;
#pragma unroll
      for (int j = 0; j < 8; ++j) {
        int k0r = imin(kb + 2 * j + half * 16, TT - 1);
        int k1r = imin(kb + 2 * j + 1 + half * 16, TT - 1);
        unsigned int lo = Vb[base + (size_t)k0r * DHD + dh];
        unsigned int hi = Vb[base + (size_t)k1r * DHD + dh];
        t[j] = lo | (hi << 16);
      }
      v16bf vf = __builtin_bit_cast(v16bf, t);
      o[nt] = __builtin_amdgcn_wmma_f32_16x16x32_bf16(false, pf, false, vf,
                                                      (short)0, o[nt], false, false);
    }
  }

  // normalize and store to (B,T,D) bf16
  const int b_ = bh >> 4, h_ = bh & 15;
#pragma unroll
  for (int r = 0; r < 8; ++r) {
    float inv = 1.0f / lrow[r];
    int row = qb * 16 + r + 8 * half;
#pragma unroll
    for (int nt = 0; nt < 4; ++nt) {
      int d = h_ * 64 + nt * 16 + l16;
      Ob[((size_t)b_ * TT + row) * DD + d] = f2bf(o[nt][r] * inv);
    }
  }
}

// ---------------- host launch ----------------
extern "C" void kernel_launch(void* const* d_in, const int* in_sizes, int n_in,
                              void* d_out, int out_size, void* d_ws, size_t ws_size,
                              hipStream_t stream) {
  const float* x  = (const float*)d_in[0];
  const float* Wq = (const float*)d_in[1];
  const float* Wk = (const float*)d_in[2];
  const float* Wv = (const float*)d_in[3];
  const float* Wo = (const float*)d_in[4];

  const size_t MiB = 1u << 20;
  char* ws = (char*)d_ws;
  unsigned short* xb    = (unsigned short*)(ws + 0);        // 8 MiB
  unsigned short* wqb   = (unsigned short*)(ws + 8 * MiB);  // 2 MiB each
  unsigned short* wkb   = (unsigned short*)(ws + 10 * MiB);
  unsigned short* wvb   = (unsigned short*)(ws + 12 * MiB);
  unsigned short* wob   = (unsigned short*)(ws + 14 * MiB);
  unsigned short* Qb    = (unsigned short*)(ws + 16 * MiB); // 8 MiB each, (B,H,T,DH)
  unsigned short* Kb    = (unsigned short*)(ws + 24 * MiB);
  unsigned short* Vb    = (unsigned short*)(ws + 32 * MiB);
  unsigned short* attnb = (unsigned short*)(ws + 40 * MiB); // 8 MiB, (B,T,D)

  // 1) fp32 -> bf16
  {
    int n4x = (BB * TT * DD) / 4;
    cvt_f32_bf16_kernel<<<(n4x + 255) / 256, 256, 0, stream>>>(x, xb, n4x);
    int n4w = (DD * DD) / 4;
    cvt_f32_bf16_kernel<<<(n4w + 255) / 256, 256, 0, stream>>>(Wq, wqb, n4w);
    cvt_f32_bf16_kernel<<<(n4w + 255) / 256, 256, 0, stream>>>(Wk, wkb, n4w);
    cvt_f32_bf16_kernel<<<(n4w + 255) / 256, 256, 0, stream>>>(Wv, wvb, n4w);
    cvt_f32_bf16_kernel<<<(n4w + 255) / 256, 256, 0, stream>>>(Wo, wob, n4w);
  }

  // 2) Q/K/V projections, head-split bf16 output
  dim3 ggrid(DD / 128, (BB * TT) / 128);
  gemm_bf16_kernel<true><<<ggrid, 256, 0, stream>>>(xb, wqb, Qb);
  gemm_bf16_kernel<true><<<ggrid, 256, 0, stream>>>(xb, wkb, Kb);
  gemm_bf16_kernel<true><<<ggrid, 256, 0, stream>>>(xb, wvb, Vb);

  // 3) flash attention: 4096 wave-tasks, 8 waves per block
  attn_kernel<<<(BB * HH * (TT / 16)) / 8, 256, 0, stream>>>(Qb, Kb, Vb, attnb);

  // 4) output projection -> fp32 d_out
  gemm_bf16_kernel<false><<<ggrid, 256, 0, stream>>>(attnb, wob, (float*)d_out);
}
